// ManyToOneDeepRNN_6399501271112
// MI455X (gfx1250) — matrix operand order repaired and checked by reading, compiled, not run on hardware
//
#include <hip/hip_runtime.h>
#include <cstdint>
#include <cstddef>

// ---------------------------------------------------------------------------
// 4-layer LSTM (many-to-one) for MI455X / gfx1250, wave32 + WMMA bf16.
//
//  * Input projections precomputed for all T as large WMMA GEMMs
//    (parallel over T*B = 16384 rows) -> fp32 preact [T*B, 4096].
//  * Recurrence: persistent kernel, 64 WGs, each owns 16 hidden units
//    (= 64 gate columns). Its W_h^T slice (64 x 1024 bf16, padded) is
//    resident in LDS for all 512 steps; h_{t-1} (64 KB bf16) is read from
//    L2 each step. Device-scope atomic barrier per timestep; next-step
//    preactivations prefetched (global_prefetch_b8) during current step.
//  * Weights converted once to bf16 and transposed (LDS-tiled, coalesced
//    both directions) so WMMA B-fragments are two contiguous 16-byte loads.
// ---------------------------------------------------------------------------

typedef __attribute__((ext_vector_type(16))) __bf16 v16bf;
typedef __attribute__((ext_vector_type(8)))  float  v8f;

#define WMMA_BF16(a, b, c) \
  __builtin_amdgcn_wmma_f32_16x16x32_bf16(false, (a), false, (b), (short)0, (c), false, false)

constexpr int B    = 32;
constexpr int T    = 512;
constexpr int DIN  = 512;
constexpr int H    = 1024;
constexpr int NL   = 4;
constexpr int OUTN = 1024;
constexpr int G4   = 4 * H;      // 4096 gate columns
constexpr int TB   = T * B;      // 16384 rows
constexpr int RG   = 64;         // recurrence grid: 64 persistent WGs
constexpr int WROW = H + 8;      // padded LDS row stride (bf16) vs bank conflicts

union Frag {
  v16bf v;
  uint4 q[2];
  __bf16 h[16];
};

__device__ __forceinline__ float sigmoidf_(float x) {
  return 1.0f / (1.0f + __expf(-x));
}

// Device-wide arrive-and-spin barrier (monotonic counter, one per layer).
__device__ __forceinline__ void grid_barrier(unsigned* ctr, unsigned target) {
  __syncthreads();
  if (threadIdx.x == 0) {
    __threadfence();  // release h stores to device scope
    __hip_atomic_fetch_add(ctr, 1u, __ATOMIC_RELEASE, __HIP_MEMORY_SCOPE_AGENT);
    while (__hip_atomic_load(ctr, __ATOMIC_ACQUIRE, __HIP_MEMORY_SCOPE_AGENT) < target) {
      __builtin_amdgcn_s_sleep(2);
    }
  }
  __syncthreads();
}

// ---------------------------------------------------------------------------
// LDS-tiled transpose + fp32->bf16:  dst[c*K + k] = W[(k0+k)*ldW + c]
// 32x32 tiles; reads and writes both coalesced. Grid: (K/32, C/32).
// ---------------------------------------------------------------------------
__global__ __launch_bounds__(256) void k_wtx(const float* __restrict__ W, int ldW,
                                             int k0, int K, int C,
                                             __bf16* __restrict__ dst) {
  __shared__ __bf16 tile[32 * 36];  // pad 36 to dodge bank conflicts
  const int tid = threadIdx.x;
  const int tk0 = blockIdx.x * 32;  // k tile base
  const int tc0 = blockIdx.y * 32;  // c tile base
#pragma unroll
  for (int it = 0; it < 4; ++it) {
    int kk = (tid >> 5) + it * 8, cc = tid & 31;     // coalesced over cc
    tile[kk * 36 + cc] = (__bf16)W[(size_t)(k0 + tk0 + kk) * ldW + tc0 + cc];
  }
  __syncthreads();
#pragma unroll
  for (int it = 0; it < 4; ++it) {
    int cc = (tid >> 5) + it * 8, kk = tid & 31;     // coalesced over kk
    dst[(size_t)(tc0 + cc) * K + tk0 + kk] = tile[kk * 36 + cc];
  }
}

// ---------------------------------------------------------------------------
// A-fragment loader (16-bit A 16x32 layout: lane holds row `row`;
// halfword e -> K = kc*32 + 16*(e/8) + 8*hi + (e%8), i.e. two 16B segments).
// ---------------------------------------------------------------------------
__device__ __forceinline__ Frag load_a_f32x(const float* __restrict__ x, int row,
                                            int kc, int hi) {
  // x laid out [B, T, DIN]; row r -> (b = r%32, t = r/32)
  Frag a;
  const float* p = x + ((size_t)(row & (B - 1)) * T + (size_t)(row >> 5)) * DIN +
                   (size_t)kc * 32 + hi * 8;
#pragma unroll
  for (int e = 0; e < 8; ++e) a.h[e] = (__bf16)p[e];
#pragma unroll
  for (int e = 0; e < 8; ++e) a.h[8 + e] = (__bf16)p[16 + e];
  return a;
}

__device__ __forceinline__ Frag load_a_bf16(const __bf16* __restrict__ hs, int ldK,
                                            int row, int kc, int hi) {
  Frag a;
  const __bf16* p = hs + (size_t)row * ldK + (size_t)kc * 32 + hi * 8;
  a.q[0] = *reinterpret_cast<const uint4*>(p);
  a.q[1] = *reinterpret_cast<const uint4*>(p + 16);
  return a;
}

// ---------------------------------------------------------------------------
// Big input-projection GEMM:  preact[r, :] = A[r, :] @ W_x   (fp32 out)
// Grid: (TB/256, 4096/64), 256 threads (8 waves).
// Each wave: 2 M-tiles x 4 N-tiles (B-fragments reused across both A-tiles),
// K-loop of 16x16x32 bf16 WMMAs -> 1.5 b128 fragment loads per WMMA.
// ---------------------------------------------------------------------------
__global__ __launch_bounds__(256) void k_input_gemm(const void* __restrict__ Asrc,
                                                    int a_is_x_f32, int K,
                                                    const __bf16* __restrict__ BT,
                                                    float* __restrict__ Cout) {
  const int lane = threadIdx.x & 31, wave = threadIdx.x >> 5;
  const int lm = lane & 15, hi = lane >> 4;
  const int mbase = blockIdx.x * 256 + wave * 32;  // 2 x 16-row M tiles
  const int n0 = blockIdx.y * 64;                  // 4 x 16-col N tiles
  const int r0 = mbase + lm, r1 = mbase + 16 + lm;

  v8f acc[2][4];
#pragma unroll
  for (int mi = 0; mi < 2; ++mi)
#pragma unroll
    for (int nt = 0; nt < 4; ++nt)
#pragma unroll
      for (int r = 0; r < 8; ++r) acc[mi][nt][r] = 0.0f;

  const int nchunk = K >> 5;
  for (int kc = 0; kc < nchunk; ++kc) {
    Frag a0, a1;
    if (a_is_x_f32) {
      a0 = load_a_f32x((const float*)Asrc, r0, kc, hi);
      a1 = load_a_f32x((const float*)Asrc, r1, kc, hi);
    } else {
      a0 = load_a_bf16((const __bf16*)Asrc, K, r0, kc, hi);
      a1 = load_a_bf16((const __bf16*)Asrc, K, r1, kc, hi);
    }
#pragma unroll
    for (int nt = 0; nt < 4; ++nt) {
      // B fragment: lane N = lm, K = kc*32 + 16*(e/8) + 8*hi + (e%8) (two 16B segments)
      Frag b;
      const __bf16* q = BT + (size_t)(n0 + nt * 16 + lm) * K + (size_t)kc * 32 + hi * 8;
      b.q[0] = *reinterpret_cast<const uint4*>(q);
      b.q[1] = *reinterpret_cast<const uint4*>(q + 16);
      acc[0][nt] = WMMA_BF16(a0.v, b.v, acc[0][nt]);
      acc[1][nt] = WMMA_BF16(a1.v, b.v, acc[1][nt]);
    }
  }
  // C layout: lane N = lm, M = 8*hi + r
#pragma unroll
  for (int mi = 0; mi < 2; ++mi)
#pragma unroll
    for (int nt = 0; nt < 4; ++nt)
#pragma unroll
      for (int r = 0; r < 8; ++r)
        Cout[(size_t)(mbase + mi * 16 + 8 * hi + r) * G4 + (size_t)(n0 + nt * 16 + lm)] =
            acc[mi][nt][r];
}

// ---------------------------------------------------------------------------
// Persistent recurrence kernel: 64 WGs x 256 threads; WG wg owns hidden units
// [wg*16, wg*16+16) i.e. gate columns {g*1024 + wg*16 + jj}.
// Wave (mt,g) computes z tile [16x16]: 32 bf16 WMMAs per step with the W_h^T
// slice resident in LDS. Gate math + c state in fp32 in LDS. Grid barrier per
// timestep publishes h_t (bf16, double-buffered in global).
// ---------------------------------------------------------------------------
__global__ __launch_bounds__(256) void k_lstm_recur(
    const float* __restrict__ precomp,   // [T*B, 4096] fp32
    const __bf16* __restrict__ WhT,      // [4096, 1024] bf16 (W_h transposed)
    const float* __restrict__ bias,      // [4096]
    const float* __restrict__ h0s,       // [32, 1024] fp32 (this layer)
    const float* __restrict__ c0s,       // [32, 1024] fp32 (this layer)
    __bf16* __restrict__ h_ring,         // [2, 32, 1024] bf16
    __bf16* __restrict__ hs_out,         // [T, 32, 1024] bf16 time-major
    unsigned* __restrict__ bar) {
  extern __shared__ char smem[];
  __bf16* lds_w = (__bf16*)smem;                                      // [64][WROW]
  float* lds_z = (float*)(smem + (size_t)64 * WROW * sizeof(__bf16)); // [32][64]
  float* lds_c = lds_z + 32 * 64;                                     // [32][16]
  float* lds_b = lds_c + 32 * 16;                                     // [64]

  const int wg = blockIdx.x;
  const int j0 = wg * 16;
  const int tid = threadIdx.x;
  const int lane = tid & 31, wave = tid >> 5;
  const int lm = lane & 15, hi = lane >> 4;
  const int mt = wave >> 2, g = wave & 3;   // 2 M-tiles x 4 gates = 8 waves

  // --- one-time: stage this WG's W_h^T slice into LDS (64 rows x 1024 bf16)
  for (int i = tid; i < 64 * (H / 8); i += 256) {   // 16B (8 bf16) chunks
    int rr = i >> 7, c8 = i & 127;
    const __bf16* src =
        WhT + (size_t)((rr >> 4) * H + j0 + (rr & 15)) * H + (size_t)c8 * 8;
    *reinterpret_cast<uint4*>(lds_w + (size_t)rr * WROW + c8 * 8) =
        *reinterpret_cast<const uint4*>(src);
  }
  for (int i = tid; i < 64; i += 256)
    lds_b[i] = bias[(i >> 4) * H + j0 + (i & 15)];
  for (int i = tid; i < 512; i += 256) {
    int row = i >> 4, jj = i & 15;
    lds_c[i] = c0s[(size_t)row * H + j0 + jj];
    h_ring[(size_t)row * H + j0 + jj] = (__bf16)h0s[(size_t)row * H + j0 + jj];
  }
  grid_barrier(bar, (unsigned)RG * 1u);  // h_ring[0] fully initialized

  const int ncol = g * 1024 + j0 + lm;   // this lane's gate column (C layout N)
  for (int t = 0; t < T; ++t) {
    const __bf16* hp = h_ring + (size_t)(t & 1) * B * H;
    __bf16* hn = h_ring + (size_t)((t + 1) & 1) * B * H;

    // accumulator = precomputed input projection + bias (C layout)
    const float bcol = lds_b[g * 16 + lm];
    v8f acc;
#pragma unroll
    for (int r = 0; r < 8; ++r)
      acc[r] = precomp[((size_t)t * B + mt * 16 + 8 * hi + r) * G4 + ncol] + bcol;

    // warm L2/L0 for next step's preactivation rows while we compute this one
    if (t + 1 < T) {
#pragma unroll
      for (int r = 0; r < 8; ++r)
        __builtin_prefetch(&precomp[((size_t)(t + 1) * B + mt * 16 + 8 * hi + r) * G4 + ncol],
                           0, 3);
    }

    // z += h_{t-1} @ W_h (slice): K = 1024, 32 chunks of 32
#pragma unroll 4
    for (int kc = 0; kc < H / 32; ++kc) {
      Frag a;
      const __bf16* ap = hp + (size_t)(mt * 16 + lm) * H + (size_t)kc * 32 + hi * 8;
      a.q[0] = *reinterpret_cast<const uint4*>(ap);
      a.q[1] = *reinterpret_cast<const uint4*>(ap + 16);
      Frag b;
      const __bf16* bp = lds_w + (size_t)(g * 16 + lm) * WROW + (size_t)kc * 32 + hi * 8;
      b.q[0] = *reinterpret_cast<const uint4*>(bp);
      b.q[1] = *reinterpret_cast<const uint4*>(bp + 16);
      acc = WMMA_BF16(a.v, b.v, acc);
    }

    // publish z tile to LDS: lds_z[row][g*16 + jj]
#pragma unroll
    for (int r = 0; r < 8; ++r)
      lds_z[(mt * 16 + 8 * hi + r) * 64 + g * 16 + lm] = acc[r];
    __syncthreads();

    // gate math: c' = sig(f)*c + sig(i)*tanh(g); h' = sig(o)*tanh(c')
    for (int i = tid; i < 512; i += 256) {
      int row = i >> 4, jj = i & 15;
      float zi = lds_z[row * 64 + jj];
      float zf = lds_z[row * 64 + 16 + jj];
      float zg = lds_z[row * 64 + 32 + jj];
      float zo = lds_z[row * 64 + 48 + jj];
      float cn = sigmoidf_(zf) * lds_c[i] + sigmoidf_(zi) * tanhf(zg);
      float hv = sigmoidf_(zo) * tanhf(cn);
      lds_c[i] = cn;
      __bf16 hb = (__bf16)hv;
      hn[(size_t)row * H + j0 + jj] = hb;
      hs_out[((size_t)t * B + row) * H + j0 + jj] = hb;
    }
    grid_barrier(bar, (unsigned)RG * (unsigned)(t + 2));  // h_t visible everywhere
  }
}

// ---------------------------------------------------------------------------
// Dense head: out[32,1024] = h_last @ Wout + bout (WMMA bf16, fp32 out)
// 16 WGs x 8 waves = 128 waves = 2 M-tiles x 64 N-tiles.
// ---------------------------------------------------------------------------
__global__ __launch_bounds__(256) void k_head(const __bf16* __restrict__ hl,  // [32,1024]
                                              const __bf16* __restrict__ WoT, // [1024,1024]
                                              const float* __restrict__ bout,
                                              float* __restrict__ out) {
  const int lane = threadIdx.x & 31, wave = threadIdx.x >> 5;
  const int lm = lane & 15, hi = lane >> 4;
  const int gw = blockIdx.x * 8 + wave;
  const int mt = gw >> 6, nt = gw & 63;
  const int n = nt * 16 + lm;

  v8f acc;
  const float bv = bout[n];
#pragma unroll
  for (int r = 0; r < 8; ++r) acc[r] = bv;

  for (int kc = 0; kc < H / 32; ++kc) {
    Frag a = load_a_bf16(hl, H, mt * 16 + lm, kc, hi);
    Frag b;
    const __bf16* bp = WoT + (size_t)n * H + (size_t)kc * 32 + hi * 8;
    b.q[0] = *reinterpret_cast<const uint4*>(bp);
    b.q[1] = *reinterpret_cast<const uint4*>(bp + 16);
    acc = WMMA_BF16(a.v, b.v, acc);
  }
#pragma unroll
  for (int r = 0; r < 8; ++r)
    out[(size_t)(mt * 16 + 8 * hi + r) * OUTN + n] = acc[r];
}

// ---------------------------------------------------------------------------
extern "C" void kernel_launch(void* const* d_in, const int* in_sizes, int n_in,
                              void* d_out, int out_size, void* d_ws, size_t ws_size,
                              hipStream_t stream) {
  (void)in_sizes; (void)n_in; (void)out_size; (void)ws_size;
  const float* x    = (const float*)d_in[0];
  const float* h0   = (const float*)d_in[1];
  const float* c0   = (const float*)d_in[2];
  const float* W0   = (const float*)d_in[3];
  const float* b0   = (const float*)d_in[4];
  const float* Wl   = (const float*)d_in[5];
  const float* bl   = (const float*)d_in[6];
  const float* Wout = (const float*)d_in[7];
  const float* bout = (const float*)d_in[8];
  float* out = (float*)d_out;

  // Workspace carve-out (~382 MiB): bf16 weights, preact, hs ping-pong, h ring.
  char* ws = (char*)d_ws;
  size_t off = 0;
  auto carve = [&](size_t bytes) -> char* {
    char* p = ws + off;
    off = (off + bytes + 255) & ~(size_t)255;
    return p;
  };
  __bf16* WT0x = (__bf16*)carve((size_t)G4 * DIN * sizeof(__bf16));
  __bf16* WT0h = (__bf16*)carve((size_t)G4 * H * sizeof(__bf16));
  __bf16* WTx[3];
  __bf16* WTh[3];
  for (int l = 0; l < 3; ++l) {
    WTx[l] = (__bf16*)carve((size_t)G4 * H * sizeof(__bf16));
    WTh[l] = (__bf16*)carve((size_t)G4 * H * sizeof(__bf16));
  }
  __bf16* WoT  = (__bf16*)carve((size_t)OUTN * H * sizeof(__bf16));
  float*  pre  = (float*)carve((size_t)TB * G4 * sizeof(float));
  __bf16* hsA  = (__bf16*)carve((size_t)TB * H * sizeof(__bf16));
  __bf16* hsB  = (__bf16*)carve((size_t)TB * H * sizeof(__bf16));
  __bf16* hrng = (__bf16*)carve((size_t)2 * B * H * sizeof(__bf16));
  unsigned* bar = (unsigned*)carve(256);

  hipMemsetAsync(bar, 0, 256, stream);  // per-layer barrier counters

  // bf16 weight conversion + transpose (LDS-tiled, coalesced)
  k_wtx<<<dim3(DIN / 32, G4 / 32), 256, 0, stream>>>(W0, G4, 0, DIN, G4, WT0x);
  k_wtx<<<dim3(H / 32, G4 / 32), 256, 0, stream>>>(W0, G4, DIN, H, G4, WT0h);
  for (int l = 0; l < 3; ++l) {
    const float* Wp = Wl + (size_t)l * (2 * H) * G4;
    k_wtx<<<dim3(H / 32, G4 / 32), 256, 0, stream>>>(Wp, G4, 0, H, G4, WTx[l]);
    k_wtx<<<dim3(H / 32, G4 / 32), 256, 0, stream>>>(Wp, G4, H, H, G4, WTh[l]);
  }
  k_wtx<<<dim3(H / 32, OUTN / 32), 256, 0, stream>>>(Wout, OUTN, 0, H, OUTN, WoT);

  const size_t smem = (size_t)64 * WROW * sizeof(__bf16) +
                      (32 * 64 + 32 * 16 + 64) * sizeof(float);  // ~139 KB

  for (int l = 0; l < NL; ++l) {
    // 1) input projection for all timesteps (big WMMA GEMM)
    if (l == 0) {
      k_input_gemm<<<dim3(TB / 256, G4 / 64), 256, 0, stream>>>(x, 1, DIN, WT0x, pre);
    } else {
      const __bf16* hin = (l == 2) ? hsB : hsA;  // l=1:A, l=2:B, l=3:A
      k_input_gemm<<<dim3(TB / 256, G4 / 64), 256, 0, stream>>>(hin, 0, H, WTx[l - 1], pre);
    }
    // 2) recurrence (persistent, grid-barrier per timestep)
    const __bf16* Wh  = (l == 0) ? WT0h : WTh[l - 1];
    const float* bias = (l == 0) ? b0 : (bl + (size_t)(l - 1) * G4);
    __bf16* hso = (l & 1) ? hsB : hsA;           // l0->A, l1->B, l2->A, l3->B
    k_lstm_recur<<<RG, 256, smem, stream>>>(pre, Wh, bias,
                                            h0 + (size_t)l * B * H,
                                            c0 + (size_t)l * B * H,
                                            hrng, hso, bar + l * 16);
  }

  // 3) dense head on h_last (= hs of layer 3 at t = T-1)
  k_head<<<16, 256, 0, stream>>>(hsB + (size_t)(T - 1) * B * H, WoT, bout, out);
}
